// MistralAttention_24713241821807
// MI455X (gfx1250) — compile-verified, matrix operand order checked by reading, not run on hardware
//
#include <hip/hip_runtime.h>
#include <hip/hip_bf16.h>

typedef __attribute__((ext_vector_type(16))) _Float16 v16h;
typedef __attribute__((ext_vector_type(8)))  _Float16 v8h;
typedef __attribute__((ext_vector_type(8)))  float    v8f;
typedef __attribute__((ext_vector_type(4)))  float    v4f;

#define S_LEN   2048
#define HDIM    4096
#define NHEADS  32
#define NKVH    8
#define HEADD   128
#define QDIM    4096   // NHEADS*HEADD
#define KVDIM   1024   // NKVH*HEADD
#define SCALE   0.08838834764831845f   // 128^-0.5
#define NEGINF  -1000000000.0f

// ---------------------------------------------------------------- helpers
__device__ __forceinline__ v16h make_v16(v8h lo, v8h hi) {
    v16h r;
#pragma unroll
    for (int i = 0; i < 8; ++i) { r[i] = lo[i]; r[i + 8] = hi[i]; }
    return r;
}

__device__ __forceinline__ v8f wmma_f16(v16h a, v16h b, v8f c) {
    // v_wmma_f32_16x16x32_f16: D = A(16x32) x B(32x16) + C(16x16 f32)
    return __builtin_amdgcn_wmma_f32_16x16x32_f16(
        false, a, false, b, (short)0, c, false, false);
}

// Async global->LDS copy (CDNA5): 16B per lane, INST_OFFSET applies to both sides.
__device__ __forceinline__ void async_copy16(unsigned lds_addr,
                                             unsigned long long gaddr) {
    asm volatile("global_load_async_to_lds_b128 %0, %1, off"
                 :: "v"(lds_addr), "v"(gaddr) : "memory");
}
__device__ __forceinline__ void async_copy16_off16(unsigned lds_addr,
                                                   unsigned long long gaddr) {
    asm volatile("global_load_async_to_lds_b128 %0, %1, off offset:16"
                 :: "v"(lds_addr), "v"(gaddr) : "memory");
}
__device__ __forceinline__ void wait_async0() {
    asm volatile("s_wait_asynccnt 0x0" ::: "memory");
}

// ---------------------------------------------------------------- fp32 -> fp16
__global__ void cvt_f32_to_f16(const float* __restrict__ src,
                               _Float16* __restrict__ dst, int n4) {
    int i = blockIdx.x * blockDim.x + threadIdx.x;
    if (i < n4) {
        float4 v = ((const float4*)src)[i];
        union { _Float16 h[4]; uint2 u; } o;
        o.h[0] = (_Float16)v.x; o.h[1] = (_Float16)v.y;
        o.h[2] = (_Float16)v.z; o.h[3] = (_Float16)v.w;
        ((uint2*)dst)[i] = o.u;
    }
}

// ---------------------------------------------------------------- GEMM  C = A * W^T
// A: MxK f16 row-major, W: NxK f16 row-major, C: MxN (fp32 or fp16)
// 128x128 tile, K-step 32, 8 waves x (32x64). Double-buffered LDS filled by
// global_load_async_to_lds_b128 (ASYNCcnt), overlapped with WMMA on other buffer.
template <typename OutT>
__global__ __launch_bounds__(256)
void gemm_abT(const _Float16* __restrict__ A, const _Float16* __restrict__ W,
              OutT* __restrict__ C, int M, int N, int K) {
    __shared__ _Float16 As[2][128][40];
    __shared__ _Float16 Bs[2][128][40];

    const int t    = threadIdx.x;
    const int w    = __builtin_amdgcn_readfirstlane(t >> 5);   // uniform wave id
    const int lane = t & 31;
    const int r    = lane & 15;
    const int hs   = lane >> 4;
    const int mbase = blockIdx.y * 128;
    const int nbase = blockIdx.x * 128;
    const int wm = (w >> 1) * 32;
    const int wn = (w & 1) * 64;

    v8f acc[2][4];
#pragma unroll
    for (int a = 0; a < 2; ++a)
#pragma unroll
        for (int b = 0; b < 4; ++b)
#pragma unroll
            for (int j = 0; j < 8; ++j) acc[a][b][j] = 0.0f;

    const int lrow = t >> 1;              // 0..127
    const int lseg = (t & 1) * 16;        // 0 or 16 halfs
    const _Float16* ga = A + (size_t)(mbase + lrow) * K + lseg;
    const _Float16* gb = W + (size_t)(nbase + lrow) * K + lseg;

    auto issue = [&](int p, int k0) {
        unsigned la = (unsigned)(uintptr_t)&As[p][lrow][lseg];
        unsigned lb = (unsigned)(uintptr_t)&Bs[p][lrow][lseg];
        unsigned long long a0 = (unsigned long long)(uintptr_t)(ga + k0);
        unsigned long long b0 = (unsigned long long)(uintptr_t)(gb + k0);
        async_copy16(la, a0);
        async_copy16_off16(la, a0);
        async_copy16(lb, b0);
        async_copy16_off16(lb, b0);
    };

    issue(0, 0);
    wait_async0();
    __syncthreads();

    int p = 0;
    const int kb = hs * 8;
#pragma unroll 1
    for (int k0 = 0; k0 < K; k0 += 32) {
        if (k0 + 32 < K) issue(p ^ 1, k0 + 32);   // prefetch next tile into LDS

        v16h af[2], bf[4];
#pragma unroll
        for (int fm = 0; fm < 2; ++fm) {
            int row = wm + fm * 16 + r;
            af[fm] = make_v16(*(const v8h*)&As[p][row][kb],
                              *(const v8h*)&As[p][row][kb + 16]);
        }
#pragma unroll
        for (int fn = 0; fn < 4; ++fn) {
            int row = wn + fn * 16 + r;
            bf[fn] = make_v16(*(const v8h*)&Bs[p][row][kb],
                              *(const v8h*)&Bs[p][row][kb + 16]);
        }
#pragma unroll
        for (int fm = 0; fm < 2; ++fm)
#pragma unroll
            for (int fn = 0; fn < 4; ++fn)
                acc[fm][fn] = wmma_f16(af[fm], bf[fn], acc[fm][fn]);

        wait_async0();     // next buffer resident
        __syncthreads();   // all waves done reading current buffer
        p ^= 1;
    }

#pragma unroll
    for (int fm = 0; fm < 2; ++fm)
#pragma unroll
        for (int fn = 0; fn < 4; ++fn) {
            int col = nbase + wn + fn * 16 + r;
#pragma unroll
            for (int j = 0; j < 8; ++j) {
                int row = mbase + wm + fm * 16 + hs * 8 + j;
                C[(size_t)row * N + col] = (OutT)acc[fm][fn][j];
            }
        }
}

// ---------------------------------------------------------------- RoPE + repack
__global__ void rope_pack(const _Float16* __restrict__ qraw,
                          const _Float16* __restrict__ kraw,
                          const _Float16* __restrict__ vraw,
                          const float* __restrict__ cosb,
                          const float* __restrict__ sinb,
                          _Float16* __restrict__ q16,
                          _Float16* __restrict__ k16,
                          _Float16* __restrict__ vT16) {
    const int d = threadIdx.x;   // 0..127
    const int s = blockIdx.x;    // 0..2047
    const int h = blockIdx.y;    // 0..31
    const float c  = cosb[s * HEADD + d];
    const float sn = sinb[s * HEADD + d];
    {
        const size_t base = (size_t)s * QDIM + h * HEADD;
        float x = (float)qraw[base + d];
        float o = (d < 64) ? -(float)qraw[base + d + 64]
                           :  (float)qraw[base + d - 64];
        q16[((size_t)h * S_LEN + s) * HEADD + d] = (_Float16)(x * c + o * sn);
    }
    if (h < NKVH) {
        const size_t base = (size_t)s * KVDIM + h * HEADD;
        float x = (float)kraw[base + d];
        float o = (d < 64) ? -(float)kraw[base + d + 64]
                           :  (float)kraw[base + d - 64];
        k16[((size_t)h * S_LEN + s) * HEADD + d] = (_Float16)(x * c + o * sn);
        vT16[((size_t)h * HEADD + d) * S_LEN + s] = vraw[base + d];
    }
}

// ---------------------------------------------------------------- fused attention
// Block = 16 query rows x one head. Scores (WMMA, causal-pruned) -> LDS ->
// exact softmax (wave-private rows, shfl reductions) -> NT-store attn_weights ->
// PV (WMMA, k-loop truncated at diagonal) -> fp16 attn_out.
__global__ __launch_bounds__(256)
void attn_kernel(const _Float16* __restrict__ q16,
                 const _Float16* __restrict__ k16,
                 const _Float16* __restrict__ vT16,
                 float* __restrict__ attw,
                 _Float16* __restrict__ ao) {
    __shared__ float sc[16][2052];     // score/prob rows, 16B-aligned pitch

    const int t     = threadIdx.x;
    const int w     = __builtin_amdgcn_readfirstlane(t >> 5);  // uniform wave id
    const int lane  = t & 31;
    const int r     = lane & 15;
    const int hs    = lane >> 4;
    const int qbase = blockIdx.x * 16;
    const int h     = blockIdx.y;
    const int kvh   = h >> 2;          // GQA: 4 query heads per KV head
    const int qlast = qbase + 15;
    const int jlim  = qbase + 16;      // causal frontier (exclusive)

    // --- Q fragments for all 4 K-steps (d = 0..127)
    v16h aq[4];
    {
        const _Float16* qp = q16 + ((size_t)h * S_LEN + qbase + r) * HEADD;
#pragma unroll
        for (int ks = 0; ks < 4; ++ks) {
            int kb = ks * 32 + hs * 8;
            aq[ks] = make_v16(*(const v8h*)(qp + kb), *(const v8h*)(qp + kb + 16));
        }
    }

    // --- scores: wave w owns keys [w*256, w*256+256); only unmasked tiles
    {
        const int rel = qlast - w * 256;
        const int nc  = (rel < 0) ? 0 : ((rel >> 4) + 1 < 16 ? (rel >> 4) + 1 : 16);
        const _Float16* kbase_p =
            k16 + ((size_t)kvh * S_LEN + w * 256 + r) * HEADD;

        v16h bk0[4], bk1[4];
        auto loadK = [&](int kt, v16h* bk) {
            const _Float16* kp = kbase_p + (size_t)kt * 16 * HEADD;
#pragma unroll
            for (int ks = 0; ks < 4; ++ks) {
                int kb = ks * 32 + hs * 8;
                bk[ks] = make_v16(*(const v8h*)(kp + kb),
                                  *(const v8h*)(kp + kb + 16));
            }
        };
        if (nc > 0) loadK(0, bk0);
#pragma unroll 1
        for (int kt = 0; kt < nc; ++kt) {
            if (kt + 1 < nc) loadK(kt + 1, bk1);   // reg double-buffer prefetch
            v8f acc;
#pragma unroll
            for (int j = 0; j < 8; ++j) acc[j] = 0.0f;
#pragma unroll
            for (int ks = 0; ks < 4; ++ks) acc = wmma_f16(aq[ks], bk0[ks], acc);

            const int col = w * 256 + kt * 16 + r;
#pragma unroll
            for (int j = 0; j < 8; ++j) {
                int qr = hs * 8 + j;
                sc[qr][col] = (col <= qbase + qr) ? acc[j] * SCALE : NEGINF;
            }
#pragma unroll
            for (int ks = 0; ks < 4; ++ks) bk0[ks] = bk1[ks];
        }
    }
    __syncthreads();

    // --- exact softmax: wave w owns rows {2w, 2w+1}; float4 lanes, shfl reduce
    const int lane4 = lane * 4;
#pragma unroll 1
    for (int rr = 0; rr < 2; ++rr) {
        const int row = 2 * w + rr;
        float m = -3.4e38f;
        for (int j = lane4; j < jlim; j += 128) {
            v4f v = *(const v4f*)&sc[row][j];
            m = fmaxf(fmaxf(fmaxf(m, v[0]), v[1]), fmaxf(v[2], v[3]));
        }
#pragma unroll
        for (int off = 16; off > 0; off >>= 1) m = fmaxf(m, __shfl_xor(m, off, 32));

        float ssum = 0.0f;
        for (int j = lane4; j < jlim; j += 128) {
            v4f v = *(const v4f*)&sc[row][j];
            v4f e;
#pragma unroll
            for (int c = 0; c < 4; ++c) e[c] = __expf(v[c] - m);  // masked -> 0
            *(v4f*)&sc[row][j] = e;
            ssum += (e[0] + e[1]) + (e[2] + e[3]);
        }
#pragma unroll
        for (int off = 16; off > 0; off >>= 1) ssum += __shfl_xor(ssum, off, 32);
        const float inv = 1.0f / ssum;

        float* outw = attw + ((size_t)h * S_LEN + qbase + row) * S_LEN;
        for (int j = lane4; j < jlim; j += 128) {
            v4f v = *(const v4f*)&sc[row][j];
            v4f pv;
#pragma unroll
            for (int c = 0; c < 4; ++c) pv[c] = v[c] * inv;
            *(v4f*)&sc[row][j] = pv;                         // probs for PV
            __builtin_nontemporal_store(pv, (v4f*)(outw + j));
        }
        v4f z = {0.0f, 0.0f, 0.0f, 0.0f};
        for (int j = jlim + lane4; j < S_LEN; j += 128)      // exact zeros
            __builtin_nontemporal_store(z, (v4f*)(outw + j));
    }
    // pad probs to next 32-col boundary so last PV tile reads zeros, not garbage
    if ((jlim & 31) != 0 && lane < 16) {
        sc[2 * w][jlim + lane]     = 0.0f;
        sc[2 * w + 1][jlim + lane] = 0.0f;
    }
    __syncthreads();

    // --- PV: wave w computes d-slice [w*16, w*16+16); truncate at diagonal
    const int db = w * 16;
    v8f acc;
#pragma unroll
    for (int j = 0; j < 8; ++j) acc[j] = 0.0f;
    const _Float16* vp = vT16 + ((size_t)kvh * HEADD + db + r) * S_LEN;
    const int kb = hs * 8;
#pragma unroll 1
    for (int nb = 0; nb < jlim; nb += 32) {
        const float* p0 = &sc[r][nb + kb];
        const float* p1 = &sc[r][nb + kb + 16];
        v4f f0 = *(const v4f*)p0, f1 = *(const v4f*)(p0 + 4);
        v4f f2 = *(const v4f*)p1, f3 = *(const v4f*)(p1 + 4);
        v16h ap;
#pragma unroll
        for (int x = 0; x < 4; ++x) {
            ap[x]      = (_Float16)f0[x];
            ap[4 + x]  = (_Float16)f1[x];
            ap[8 + x]  = (_Float16)f2[x];
            ap[12 + x] = (_Float16)f3[x];
        }
        v16h bp = make_v16(*(const v8h*)(vp + nb + kb),
                           *(const v8h*)(vp + nb + kb + 16));
        acc = wmma_f16(ap, bp, acc);
    }
#pragma unroll
    for (int j = 0; j < 8; ++j) {
        int row = qbase + hs * 8 + j;
        ao[(size_t)row * QDIM + h * HEADD + db + r] = (_Float16)acc[j];
    }
}

// ---------------------------------------------------------------- launch
extern "C" void kernel_launch(void* const* d_in, const int* in_sizes, int n_in,
                              void* d_out, int out_size, void* d_ws, size_t ws_size,
                              hipStream_t stream) {
    const float* hidden = (const float*)d_in[0];   // (1,2048,4096)
    const float* wq     = (const float*)d_in[1];   // (4096,4096)
    const float* wk     = (const float*)d_in[2];   // (1024,4096)
    const float* wv     = (const float*)d_in[3];   // (1024,4096)
    const float* wo     = (const float*)d_in[4];   // (4096,4096)
    const float* cosb   = (const float*)d_in[5];   // (1,2048,128)
    const float* sinb   = (const float*)d_in[6];   // (1,2048,128)
    // d_in[7]: attn_mask — causal mask computed analytically instead.

    float* out  = (float*)d_out;                       // (1,2048,4096)
    float* attw = out + (size_t)S_LEN * HDIM;          // (1,32,2048,2048)

    char*  ws  = (char*)d_ws;
    auto alloc = [&](size_t bytes) {
        char* p = ws;
        ws += (bytes + 255) & ~(size_t)255;
        return p;
    };
    _Float16* hid16 = (_Float16*)alloc((size_t)S_LEN * HDIM * 2);
    _Float16* wq16  = (_Float16*)alloc((size_t)QDIM  * HDIM * 2);
    _Float16* wk16  = (_Float16*)alloc((size_t)KVDIM * HDIM * 2);
    _Float16* wv16  = (_Float16*)alloc((size_t)KVDIM * HDIM * 2);
    _Float16* wo16  = (_Float16*)alloc((size_t)HDIM  * QDIM * 2);
    _Float16* qraw  = (_Float16*)alloc((size_t)S_LEN * QDIM * 2);
    _Float16* kraw  = (_Float16*)alloc((size_t)S_LEN * KVDIM * 2);
    _Float16* vraw  = (_Float16*)alloc((size_t)S_LEN * KVDIM * 2);
    _Float16* q16   = (_Float16*)alloc((size_t)S_LEN * QDIM * 2);
    _Float16* k16   = (_Float16*)alloc((size_t)S_LEN * KVDIM * 2);
    _Float16* vT16  = (_Float16*)alloc((size_t)S_LEN * KVDIM * 2);
    _Float16* ao16  = qraw;  // reuse: qraw dead after rope_pack

    auto cvt = [&](const float* s, _Float16* d, size_t n) {
        int n4 = (int)(n / 4);
        cvt_f32_to_f16<<<(n4 + 255) / 256, 256, 0, stream>>>(s, d, n4);
    };
    cvt(hidden, hid16, (size_t)S_LEN * HDIM);
    cvt(wq,     wq16,  (size_t)QDIM  * HDIM);
    cvt(wk,     wk16,  (size_t)KVDIM * HDIM);
    cvt(wv,     wv16,  (size_t)KVDIM * HDIM);
    cvt(wo,     wo16,  (size_t)HDIM  * QDIM);

    // QKV projections (fp16 out, fp32 accumulate)
    gemm_abT<_Float16><<<dim3(QDIM / 128,  S_LEN / 128), 256, 0, stream>>>(
        hid16, wq16, qraw, S_LEN, QDIM, HDIM);
    gemm_abT<_Float16><<<dim3(KVDIM / 128, S_LEN / 128), 256, 0, stream>>>(
        hid16, wk16, kraw, S_LEN, KVDIM, HDIM);
    gemm_abT<_Float16><<<dim3(KVDIM / 128, S_LEN / 128), 256, 0, stream>>>(
        hid16, wv16, vraw, S_LEN, KVDIM, HDIM);

    // RoPE + head-major repack (+ V transpose)
    rope_pack<<<dim3(S_LEN, NHEADS), HEADD, 0, stream>>>(
        qraw, kraw, vraw, cosb, sinb, q16, k16, vT16);

    // Fused attention (writes attn_weights + attn_out)
    attn_kernel<<<dim3(S_LEN / 16, NHEADS), 256, 0, stream>>>(
        q16, k16, vT16, attw, ao16);

    // Output projection (fp32 out)
    gemm_abT<float><<<dim3(HDIM / 128, S_LEN / 128), 256, 0, stream>>>(
        ao16, wo16, out, S_LEN, HDIM, QDIM);
}
